// ContentAttention_16320875725106
// MI455X (gfx1250) — matrix operand order repaired and checked by reading, not compiled, not run on hardware
//
#include <hip/hip_runtime.h>
#include <hip/hip_bf16.h>
#include <math.h>
#include <stdint.h>

// ---------------- problem constants (from reference) ----------------
#define BB   16
#define NTOK 3136
#define CC   256
#define HH   8
#define HD   32
#define WS   49
#define OV   20
#define NW   64
#define WIMG 56
#define MROW (BB * NTOK)    // 50176
#define MT_PER_B (NTOK / 16) // 196: M-tiles per batch image (tile never crosses batch)

typedef __bf16 bf16_t;
typedef __attribute__((ext_vector_type(16))) __bf16 v16bf;
typedef __attribute__((ext_vector_type(8)))  float  v8f;

// ---------------- WMMA helpers (CDNA5 16x16x32 bf16, f32 accum) ----------------
__device__ inline v8f wmma_bf16(v16bf a, v16bf b, v8f c) {
  return __builtin_amdgcn_wmma_f32_16x16x32_bf16(false, a, false, b, (short)0, c,
                                                 false, false);
}

// A fragment: 16x32 (MxK) row-major source, element (r,k) = base[r*lda + k]
__device__ inline v16bf frag_a_rm(const bf16_t* base, int lda, int lane) {
  int r = lane & 15, hi = lane >> 4;
  v16bf f;
#pragma unroll
  for (int v = 0; v < 8; ++v) {
    int kb = 16 * hi + 2 * v;
    f[2 * v]     = base[r * lda + kb];
    f[2 * v + 1] = base[r * lda + kb + 1];
  }
  return f;
}

// B fragment: 32x16 (KxN); lane 0-15: col N=lane, K=0..15; lane 16-31: K=16..31
// Source stored N x K row-major: element (k,n) = base[n*ldb + k]
__device__ inline v16bf frag_b_nk(const bf16_t* base, int ldb, int lane) {
  int nn = lane & 15;
  int kb = (lane >> 4) * 16;
  const bf16_t* src = base + nn * ldb + kb;   // 16 contiguous bf16 = 32B load
  v16bf f;
#pragma unroll
  for (int j = 0; j < 16; ++j) f[j] = src[j];
  return f;
}

// Source stored K x N row-major: element (k,n) = base[k*ldb + n]
__device__ inline v16bf frag_b_kn(const bf16_t* base, int ldb, int lane) {
  int nn = lane & 15;
  int kb = (lane >> 4) * 16;
  v16bf f;
#pragma unroll
  for (int j = 0; j < 16; ++j) f[j] = base[(kb + j) * ldb + nn];
  return f;
}

// CDNA5 async global->LDS copy (ASYNCcnt path, bypasses VGPR writeback).
// GVS mode: mem = SADDR(64) + VADDR(32); LDS dest = VDST(32-bit LDS byte addr).
__device__ inline void async_b128(unsigned lds_addr, unsigned goff,
                                  unsigned long long base) {
  asm volatile("global_load_async_to_lds_b128 %0, %1, %2"
               :: "v"(lds_addr), "v"(goff), "s"(base) : "memory");
}
__device__ inline void wait_async0() {
  asm volatile("s_wait_asynccnt 0x0" ::: "memory");
}
__device__ inline unsigned lds_lo32(const void* p) {
  // generic pointer to LDS: low 32 bits hold the workgroup-relative LDS address
  return (unsigned)(uintptr_t)p;
}

// ---------------- kernels ----------------

__global__ __launch_bounds__(256) void k_cast_bf16(const float* __restrict__ in,
                                                   bf16_t* __restrict__ out, int n) {
  int i = blockIdx.x * blockDim.x + threadIdx.x;
  if (i < n) out[i] = (bf16_t)in[i];
}

// qkv = x @ w_qkv^T + b_qkv, scattered into layout qkv[s][b][h][n][d].
// Writes bf16 copy (attention operands) + fp32 copy for s=0 (clustering) and
// s=2 (conv branch); fp32 K is never consumed.
__global__ __launch_bounds__(256) void k_qkv_gemm(const bf16_t* __restrict__ xb,
                                                  const bf16_t* __restrict__ wb,
                                                  const float* __restrict__ bias,
                                                  float* __restrict__ qkvf,
                                                  bf16_t* __restrict__ qkvb) {
  int wave = threadIdx.x >> 5, lane = threadIdx.x & 31;
  int mt = blockIdx.x * 8 + wave;   // 0..3135
  int nt = blockIdx.y;              // 0..47
  const bf16_t* arow  = xb + (size_t)(mt * 16) * CC;
  const bf16_t* bbase = wb + (size_t)(nt * 16) * CC;
  v8f acc = {};
#pragma unroll
  for (int kk = 0; kk < CC; kk += 32) {
    __builtin_prefetch(arow + kk + 64, 0, 0);    // global_prefetch_b8
    v16bf a  = frag_a_rm(arow + kk, CC, lane);
    v16bf bb = frag_b_nk(bbase + kk, CC, lane);
    acc = wmma_bf16(a, bb, acc);
  }
  int nn = lane & 15, hi = lane >> 4;
  int col = nt * 16 + nn;
  int s = col >> 8, rem = col & 255, h = rem >> 5, d = rem & 31;
  int bidx = mt / MT_PER_B;                 // one division per thread (hoisted)
  int tb = (mt - bidx * MT_PER_B) * 16 + 8 * hi;
  float bv = bias[col];
  size_t rowbase = (((size_t)s * BB + bidx) * HH + h) * NTOK;
#pragma unroll
  for (int i = 0; i < 8; ++i) {
    size_t o = (rowbase + tb + i) * HD + d;
    float val = acc[i] + bv;
    qkvb[o] = (bf16_t)val;
    if (s != 1) qkvf[o] = val;
  }
}

// q_pre[bh][d][n] = qkv[0][b][h][n][d];  idx[bh][n] = n
// grid = ((NTOK+255)/256, 128*32); blockIdx.y = bh*32 + d
__global__ __launch_bounds__(256) void k_init_cluster(const float* __restrict__ qkvf,
                                                      float* __restrict__ qpre,
                                                      int* __restrict__ idx) {
  int n = blockIdx.x * blockDim.x + threadIdx.x;
  if (n >= NTOK) return;
  int bhd = blockIdx.y;
  int bh = bhd >> 5, d = bhd & 31;
  qpre[(size_t)bhd * NTOK + n] = qkvf[((size_t)bh * NTOK + n) * HD + d];
  if (d == 0) idx[(size_t)bh * NTOK + n] = n;
}

// One soft-clustering split level. One workgroup per sub-problem p of (32, n).
__global__ __launch_bounds__(256) void k_cluster_level(const float* __restrict__ qin,
                                                       float* __restrict__ qout,
                                                       const int* __restrict__ idxin,
                                                       int* __restrict__ idxout,
                                                       int n, int npad) {
  __shared__ float s_key[4096];
  __shared__ int   s_val[4096];
  __shared__ float s_coln[NTOK];
  __shared__ float s_avg[2][32];
  __shared__ float s_red[512];
  float* soft0 = s_key;                          // alias: sort buffers reused
  float* soft1 = reinterpret_cast<float*>(s_val);

  int p = blockIdx.x, tid = threadIdx.x;
  const float* Q = qin + (size_t)p * 32 * n;     // Q[d*n + j]
  int half = n >> 1;

  // 1) initial q_avg = l2norm(mean of halves)
  if (tid < 64) {
    int d = tid & 31, c = tid >> 5;
    const float* row = Q + (size_t)d * n + c * half;
    float s = 0.f;
    for (int j = 0; j < half; ++j) s += row[j];
    s_avg[c][d] = s / (float)half;
  }
  __syncthreads();
  if (tid < 2) {
    float s = 0.f;
    for (int d = 0; d < 32; ++d) { float v = s_avg[tid][d]; s += v * v; }
    s_red[tid] = 1.0f / fmaxf(sqrtf(s), 1e-12f);
  }
  __syncthreads();
  if (tid < 64) { int d = tid & 31, c = tid >> 5; s_avg[c][d] *= s_red[c]; }
  __syncthreads();

  // 2) reciprocal column norms (qn)
  for (int j = tid; j < n; j += 256) {
    float s = 0.f;
    for (int d = 0; d < 32; ++d) { float v = Q[(size_t)d * n + j]; s += v * v; }
    s_coln[j] = 1.0f / fmaxf(sqrtf(s), 1e-12f);
  }
  __syncthreads();

  // 3) two refinement iterations
  for (int it = 0; it < 2; ++it) {
    for (int j = tid; j < n; j += 256) {
      float d0 = 0.f, d1 = 0.f;
      for (int d = 0; d < 32; ++d) {
        float v = Q[(size_t)d * n + j];
        d0 += v * s_avg[0][d];
        d1 += v * s_avg[1][d];
      }
      float r = s_coln[j];
      float sc0 = d0 * r * 100.f, sc1 = d1 * r * 100.f;
      float m = fmaxf(sc0, sc1);
      float e0 = __expf(sc0 - m), e1 = __expf(sc1 - m);
      float inv = 1.f / (e0 + e1);
      soft0[j] = e0 * inv;
      soft1[j] = e1 * inv;
    }
    __syncthreads();
    {
      int d = tid & 31, grp = tid >> 5;   // 8 j-groups
      float a0 = 0.f, a1 = 0.f;
      for (int j = grp; j < n; j += 8) {
        float v = Q[(size_t)d * n + j];
        a0 += v * soft0[j];
        a1 += v * soft1[j];
      }
      s_red[tid] = a0;
      s_red[256 + tid] = a1;
    }
    __syncthreads();
    if (tid < 64) {
      int d = tid & 31, c = tid >> 5;
      float s = 0.f;
      for (int g = 0; g < 8; ++g) s += s_red[c * 256 + (g << 5) + d];
      s_avg[c][d] = s;
    }
    __syncthreads();
    if (tid < 2) {
      float s = 0.f;
      for (int d = 0; d < 32; ++d) { float v = s_avg[tid][d]; s += v * v; }
      s_red[tid] = 1.0f / fmaxf(sqrtf(s), 1e-12f);
    }
    __syncthreads();
    if (tid < 64) { int d = tid & 31, c = tid >> 5; s_avg[c][d] *= s_red[c]; }
    __syncthreads();
  }

  // 4) ratio keys (+index values; pad with +inf)
  for (int j = tid; j < npad; j += 256) {
    if (j < n) {
      float d0 = 0.f, d1 = 0.f;
      for (int d = 0; d < 32; ++d) {
        float v = Q[(size_t)d * n + j];
        d0 += v * s_avg[0][d];
        d1 += v * s_avg[1][d];
      }
      float r = s_coln[j];
      s_key[j] = (d0 * r + 1.f) / (d1 * r + 1.f);
      s_val[j] = j;
    } else {
      s_key[j] = 3.4e38f;
      s_val[j] = 0x7fffffff;
    }
  }
  __syncthreads();

  // 5) bitonic argsort, stable via (key, idx) lexicographic compare
  for (int k = 2; k <= npad; k <<= 1) {
    for (int j2 = k >> 1; j2 > 0; j2 >>= 1) {
      for (int i = tid; i < npad; i += 256) {
        int ix = i ^ j2;
        if (ix > i) {
          float ka = s_key[i], kb = s_key[ix];
          int va = s_val[i], vb = s_val[ix];
          bool up = ((i & k) == 0);
          bool agtb = (ka > kb) || (ka == kb && va > vb);
          if (agtb == up) {
            s_key[i] = kb; s_key[ix] = ka;
            s_val[i] = vb; s_val[ix] = va;
          }
        }
      }
      __syncthreads();
    }
  }

  // 6) apply permutation: running index gather + q_pre gather/split
  const int* Iin = idxin + (size_t)p * n;
  int* Iout = idxout + (size_t)p * n;
  for (int j = tid; j < n; j += 256) Iout[j] = Iin[s_val[j]];
  float* O0 = qout + (size_t)(2 * p) * 32 * half;
  float* O1 = O0 + (size_t)32 * half;
  for (int u = tid; u < 32 * half; u += 256) {
    int d = u / half, j = u % half;
    O0[(size_t)d * half + j] = Q[(size_t)d * n + s_val[j]];
    O1[(size_t)d * half + j] = Q[(size_t)d * n + s_val[half + j]];
  }
}

// Windowed attention with overlap. One workgroup (128 thr, 4 waves) per (b,wi,h).
// Q/K/V rows are gathered straight into LDS via the CDNA5 async pipe.
__global__ __launch_bounds__(128) void k_attn(const bf16_t* __restrict__ qkvb,
                                              const int* __restrict__ qidx,
                                              float* __restrict__ xatt) {
  __shared__ bf16_t q_s[64 * 32];
  __shared__ bf16_t k_s[80 * 32];
  __shared__ bf16_t v_s[96 * 32];
  __shared__ float  s_s[64 * 80];
  __shared__ bf16_t p_s[64 * 96];

  int tid = threadIdx.x;
  int h  = blockIdx.x % HH;
  int wi = (blockIdx.x / HH) % NW;
  int b  = blockIdx.x / (HH * NW);
  const int* idxrow = qidx + (size_t)(b * HH + h) * NTOK;
  const size_t headoff = (size_t)(b * HH + h) * NTOK * HD;
  const size_t sstride = (size_t)BB * HH * NTOK * HD;
  unsigned long long Qb = (unsigned long long)(uintptr_t)(qkvb + headoff);
  unsigned long long Kb = (unsigned long long)(uintptr_t)(qkvb + sstride + headoff);
  unsigned long long Vb = (unsigned long long)(uintptr_t)(qkvb + 2 * sstride + headoff);
  unsigned qlds = lds_lo32(q_s), klds = lds_lo32(k_s), vlds = lds_lo32(v_s);

  // zero only the pad regions (disjoint from async targets)
  for (int u = tid; u < (64 - WS) * HD; u += 128) q_s[WS * HD + u] = (bf16_t)0.f;
  for (int u = tid; u < (80 - (WS + OV)) * HD; u += 128) k_s[(WS + OV) * HD + u] = (bf16_t)0.f;
  for (int u = tid; u < (96 - (WS + OV)) * HD; u += 128) v_s[(WS + OV) * HD + u] = (bf16_t)0.f;

  // async gather: 64B rows as 4 x B128 per row
  for (int u = tid; u < WS * 4; u += 128) {
    int r = u >> 2, ch = (u & 3) * 16;
    int t = idxrow[wi * WS + r];
    async_b128(qlds + r * 64 + ch, (unsigned)(t * 64 + ch), Qb);
  }
  for (int u = tid; u < (WS + OV) * 4; u += 128) {
    int r = u >> 2, ch = (u & 3) * 16;
    int pos;
    if (r < WS)              pos = wi * WS + r;
    else if ((wi & 1) == 0)  pos = (wi + 1) * WS + (r - WS);
    else                     pos = (wi - 1) * WS + (WS - OV) + (r - WS);
    int t = idxrow[pos];
    async_b128(klds + r * 64 + ch, (unsigned)(t * 64 + ch), Kb);
    async_b128(vlds + r * 64 + ch, (unsigned)(t * 64 + ch), Vb);
  }
  wait_async0();
  __syncthreads();

  int wave = tid >> 5, lane = tid & 31;

  // S = (Q K^T) * scale  -- each wave one 16-row M-tile, 5 N-tiles of keys
  {
    const float scale = 0.17677669529663687f;  // 32^-0.5
    v16bf a = frag_a_rm(q_s + wave * 16 * 32, 32, lane);
    int nn = lane & 15, hi = lane >> 4;
#pragma unroll
    for (int nt = 0; nt < 5; ++nt) {
      v16bf bb = frag_b_nk(k_s + nt * 16 * 32, 32, lane);  // B(k,n)=K[n][k]
      v8f z = {};
      v8f acc = wmma_bf16(a, bb, z);
#pragma unroll
      for (int i = 0; i < 8; ++i)
        s_s[(wave * 16 + i + 8 * hi) * 80 + nt * 16 + nn] = acc[i] * scale;
    }
  }
  __syncthreads();

  // row softmax over 69 valid keys -> bf16 P (zero padded to 64x96)
  if (tid < 64) {
    int r = tid;
    if (r < WS) {
      float m = -3.4e38f;
      for (int j = 0; j < WS + OV; ++j) m = fmaxf(m, s_s[r * 80 + j]);
      float sum = 0.f;
      for (int j = 0; j < WS + OV; ++j) sum += __expf(s_s[r * 80 + j] - m);
      float inv = 1.f / sum;
      for (int j = 0; j < 96; ++j)
        p_s[r * 96 + j] =
            (j < WS + OV) ? (bf16_t)(__expf(s_s[r * 80 + j] - m) * inv) : (bf16_t)0.f;
    } else {
      for (int j = 0; j < 96; ++j) p_s[r * 96 + j] = (bf16_t)0.f;
    }
  }
  __syncthreads();

  // O = P V, scatter back through q_idx (fuses the un-permute)
  {
    v8f acc[2] = {{}, {}};
#pragma unroll
    for (int kk = 0; kk < 3; ++kk) {
      v16bf a = frag_a_rm(p_s + wave * 16 * 96 + kk * 32, 96, lane);
#pragma unroll
      for (int nt = 0; nt < 2; ++nt) {
        v16bf bb = frag_b_kn(v_s + kk * 32 * 32 + nt * 16, 32, lane);  // B(k,n)=V[k][n]
        acc[nt] = wmma_bf16(a, bb, acc[nt]);
      }
    }
    int nn = lane & 15, hi = lane >> 4;
#pragma unroll
    for (int nt = 0; nt < 2; ++nt)
#pragma unroll
      for (int i = 0; i < 8; ++i) {
        int m = wave * 16 + i + 8 * hi;
        if (m < WS) {
          int t = idxrow[wi * WS + m];
          xatt[((size_t)b * NTOK + t) * CC + h * HD + nt * 16 + nn] = acc[nt][i];
        }
      }
  }
}

// depthwise 3x3 conv on un-permuted V (identity gather/scatter cancels),
// fused add with x_att and bf16 cast for projection GEMM input.
// grid = (WIMG, BB*CC), block = 64; blockIdx.y = b*256 + c
__global__ __launch_bounds__(64) void k_dwconv_add(const float* __restrict__ qkvf,
                                                   const float* __restrict__ xatt,
                                                   const float* __restrict__ wconv,
                                                   const float* __restrict__ bconv,
                                                   bf16_t* __restrict__ a2) {
  int x = threadIdx.x;
  if (x >= WIMG) return;
  int y = blockIdx.x;
  int bc = blockIdx.y;
  int c = bc & 255, b = bc >> 8;
  int h = c >> 5, d = c & 31;
  const float* Vg = qkvf + (((size_t)2 * BB + b) * HH + h) * NTOK * HD;
  float s = bconv[c];
#pragma unroll
  for (int dy = -1; dy <= 1; ++dy) {
    int yy = y + dy;
    if (yy < 0 || yy >= WIMG) continue;
#pragma unroll
    for (int dx = -1; dx <= 1; ++dx) {
      int xx = x + dx;
      if (xx < 0 || xx >= WIMG) continue;
      s += Vg[(size_t)(yy * WIMG + xx) * HD + d] * wconv[c * 9 + (dy + 1) * 3 + (dx + 1)];
    }
  }
  size_t o = ((size_t)b * NTOK + y * WIMG + x) * CC + c;
  a2[o] = (bf16_t)(xatt[o] + s);
}

// y = (x_att + vc) @ w_proj^T + b_proj
__global__ __launch_bounds__(256) void k_proj_gemm(const bf16_t* __restrict__ a2,
                                                   const bf16_t* __restrict__ wb,
                                                   const float* __restrict__ bias,
                                                   float* __restrict__ out) {
  int wave = threadIdx.x >> 5, lane = threadIdx.x & 31;
  int mt = blockIdx.x * 8 + wave;
  int nt = blockIdx.y;
  const bf16_t* arow  = a2 + (size_t)(mt * 16) * CC;
  const bf16_t* bbase = wb + (size_t)(nt * 16) * CC;
  v8f acc = {};
#pragma unroll
  for (int kk = 0; kk < CC; kk += 32) {
    __builtin_prefetch(arow + kk + 64, 0, 0);
    v16bf a  = frag_a_rm(arow + kk, CC, lane);
    v16bf bb = frag_b_nk(bbase + kk, CC, lane);
    acc = wmma_bf16(a, bb, acc);
  }
  int nn = lane & 15, hi = lane >> 4;
  int col = nt * 16 + nn;
  float bv = bias[col];
#pragma unroll
  for (int i = 0; i < 8; ++i) {
    size_t row = (size_t)mt * 16 + i + 8 * hi;
    out[row * CC + col] = acc[i] + bv;
  }
}

// ---------------- host orchestration ----------------
extern "C" void kernel_launch(void* const* d_in, const int* in_sizes, int n_in,
                              void* d_out, int out_size, void* d_ws, size_t ws_size,
                              hipStream_t stream) {
  (void)in_sizes; (void)n_in; (void)out_size; (void)ws_size;
  const float* x      = (const float*)d_in[0];
  const float* w_qkv  = (const float*)d_in[1];
  const float* b_qkv  = (const float*)d_in[2];
  const float* w_proj = (const float*)d_in[3];
  const float* b_proj = (const float*)d_in[4];
  const float* w_conv = (const float*)d_in[5];
  const float* b_conv = (const float*)d_in[6];
  float* out = (float*)d_out;

  char* ws = (char*)d_ws;
  size_t off = 0;
  auto alloc = [&](size_t bytes) -> void* {
    void* p = ws + off;
    off += (bytes + 255) & ~(size_t)255;
    return p;
  };
  bf16_t* xb     = (bf16_t*)alloc((size_t)MROW * CC * 2);
  bf16_t* wqkvb  = (bf16_t*)alloc((size_t)3 * CC * CC * 2);
  bf16_t* wprojb = (bf16_t*)alloc((size_t)CC * CC * 2);
  float*  qkvf   = (float*) alloc((size_t)3 * MROW * CC * 4);
  bf16_t* qkvb   = (bf16_t*)alloc((size_t)3 * MROW * CC * 2);
  float*  qpreA  = (float*) alloc((size_t)128 * HD * NTOK * 4);
  float*  qpreB  = (float*) alloc((size_t)128 * HD * NTOK * 4);
  int*    idxA   = (int*)   alloc((size_t)128 * NTOK * 4);
  int*    idxB   = (int*)   alloc((size_t)128 * NTOK * 4);
  float*  xatt   = (float*) alloc((size_t)MROW * CC * 4);
  bf16_t* a2     = (bf16_t*)alloc((size_t)MROW * CC * 2);

  // 1) fp32 -> bf16 casts for WMMA operands
  k_cast_bf16<<<(MROW * CC + 255) / 256, 256, 0, stream>>>(x, xb, MROW * CC);
  k_cast_bf16<<<(3 * CC * CC + 255) / 256, 256, 0, stream>>>(w_qkv, wqkvb, 3 * CC * CC);
  k_cast_bf16<<<(CC * CC + 255) / 256, 256, 0, stream>>>(w_proj, wprojb, CC * CC);

  // 2) QKV GEMM (WMMA bf16) -> fp32 (Q,V) + bf16 (Q,K,V) copies
  {
    dim3 g(MROW / 128, (3 * CC) / 16);
    k_qkv_gemm<<<g, 256, 0, stream>>>(xb, wqkvb, b_qkv, qkvf, qkvb);
  }

  // 3) clustering sort: init + 6 split levels
  {
    dim3 g((NTOK + 255) / 256, 128 * HD);
    k_init_cluster<<<g, 256, 0, stream>>>(qkvf, qpreA, idxA);
  }
  float* qin = qpreA; float* qout = qpreB;
  int* iin = idxA;    int* iout = idxB;
  int n = NTOK;
  for (int l = 0; l < 6; ++l) {
    int bh = 128 << l;
    int npad = 1; while (npad < n) npad <<= 1;
    k_cluster_level<<<bh, 256, 0, stream>>>(qin, qout, iin, iout, n, npad);
    { float* t = qin; qin = qout; qout = t; }
    { int* t = iin; iin = iout; iout = t; }
    n >>= 1;
  }
  // final permutation now in `iin` (6 swaps -> idxA)

  // 4) windowed attention (WMMA QK^T + PV, async-LDS staging), scatters x_att
  k_attn<<<BB * NW * HH, 128, 0, stream>>>(qkvb, iin, xatt);

  // 5) depthwise conv + residual add + bf16 cast
  {
    dim3 g(WIMG, BB * CC);
    k_dwconv_add<<<g, 64, 0, stream>>>(qkvf, xatt, w_conv, b_conv, a2);
  }

  // 6) projection GEMM (WMMA bf16) -> fp32 output
  {
    dim3 g(MROW / 128, CC / 16);
    k_proj_gemm<<<g, 256, 0, stream>>>(a2, wprojb, b_proj, out);
  }
}
